// TransformerGPT2_6811818132092
// MI455X (gfx1250) — compile-verified, matrix operand order checked by reading, not run on hardware
//
#include <hip/hip_runtime.h>
#include <hip/hip_bf16.h>
#include <math.h>

typedef __attribute__((ext_vector_type(16))) _Float16 v16h;
typedef __attribute__((ext_vector_type(8)))  float    v8f;
typedef __attribute__((ext_vector_type(4)))  _Float16 v4h;

#define T_SEQ 2048
#define EMB   1024
#define NH    16
#define DHEAD 64
#define NL    8

static __device__ __forceinline__ v8f wmma_f16(v16h a, v16h b, v8f c) {
  // D = A(16x32 f16) x B(32x16 f16) + C(16x16 f32)
  return __builtin_amdgcn_wmma_f32_16x16x32_f16(false, a, false, b, (short)0, c,
                                                false, false);
}

// Build a 16-half fragment from two contiguous 16B chunks (ds_load_b128 x2).
static __device__ __forceinline__ v16h ld_frag(const _Float16* p0, const _Float16* p1) {
  v16h f;
  ((uint4*)&f)[0] = *(const uint4*)p0;
  ((uint4*)&f)[1] = *(const uint4*)p1;
  return f;
}

// Async DMA: 16 bytes global -> LDS, tracked by ASYNCcnt (GVS addressing:
// 64-bit SGPR base + 32-bit per-lane byte offset). LDS offset = low 32 bits
// of the generic pointer (ISA 10.2 aperture rule).
static __device__ __forceinline__ void async_b128(const void* gbase, int byteOff,
                                                  void* ldsDst) {
  unsigned int lds = (unsigned int)(unsigned long long)ldsDst;
  asm volatile("global_load_async_to_lds_b128 %0, %1, %2"
               :: "v"(lds), "v"(byteOff), "s"(gbase) : "memory");
}
static __device__ __forceinline__ void wait_async0() {
  asm volatile("s_wait_asynccnt 0" ::: "memory");
}

// ---------------------------------------------------------------- utilities
__global__ void copy_f32(const float* __restrict__ src, float* __restrict__ dst, int n) {
  int i = (blockIdx.x * blockDim.x + threadIdx.x) * 4;
  if (i < n) *(float4*)(dst + i) = *(const float4*)(src + i);
}

__global__ void cvt_f32_f16(const float* __restrict__ src, _Float16* __restrict__ dst, int n) {
  int i = (blockIdx.x * blockDim.x + threadIdx.x) * 4;
  if (i < n) {
    float4 v = *(const float4*)(src + i);
    v4h o;
    o[0] = (_Float16)v.x; o[1] = (_Float16)v.y;
    o[2] = (_Float16)v.z; o[3] = (_Float16)v.w;
    *(v4h*)(dst + i) = o;
  }
}

// LayerNorm over E=1024, one block (256 thr) per row, f32 stats, f16 output.
__global__ __launch_bounds__(256) void ln_f16(const float* __restrict__ h,
                                              const float* __restrict__ w,
                                              const float* __restrict__ b,
                                              _Float16* __restrict__ y) {
  __shared__ float red[512];
  const int row = blockIdx.x, t = threadIdx.x;
  const float* hr = h + (size_t)row * EMB;
  float4 v = *(const float4*)(hr + t * 4);
  red[t]       = v.x + v.y + v.z + v.w;
  red[256 + t] = v.x * v.x + v.y * v.y + v.z * v.z + v.w * v.w;
  __syncthreads();
  for (int o = 128; o > 0; o >>= 1) {
    if (t < o) { red[t] += red[t + o]; red[256 + t] += red[256 + t + o]; }
    __syncthreads();
  }
  const float mu  = red[0] * (1.0f / EMB);
  const float var = red[256] * (1.0f / EMB) - mu * mu;
  const float rs  = rsqrtf(var + 1e-5f);
  float4 wv = *(const float4*)(w + t * 4);
  float4 bv = *(const float4*)(b + t * 4);
  v4h o;
  o[0] = (_Float16)((v.x - mu) * rs * wv.x + bv.x);
  o[1] = (_Float16)((v.y - mu) * rs * wv.y + bv.y);
  o[2] = (_Float16)((v.z - mu) * rs * wv.z + bv.z);
  o[3] = (_Float16)((v.w - mu) * rs * wv.w + bv.w);
  *(v4h*)(y + (size_t)row * EMB + t * 4) = o;
}

// --------------------------------------------------------------- WMMA GEMM
// C[M,N] = A[M,K](f16) x B[K,N](f16) + bias. 128x128 block tile, 8 waves,
// each wave 64x32 (4x2 16x16 accumulators). Double-buffered LDS: next A tile
// arrives via async global->LDS DMA while WMMAs consume the current tile;
// next B tile is global-loaded to VGPRs before compute and ds-stored
// (transposed to [n][k]) after compute. MODE: 0=f16 out, 1=gelu->f16,
// 2=f32 residual add (outf = resid + acc + bias, resid may alias outf).
template<int MODE>
__global__ __launch_bounds__(256) void gemm_wmma(
    const _Float16* __restrict__ A, const _Float16* __restrict__ B,
    const float* __restrict__ bias, _Float16* out16,
    float* outf, const float* resid, int M, int N, int K) {
  __shared__ _Float16 As[2][128 * 32];   // [m][k]
  __shared__ _Float16 BsT[2][128 * 32];  // [n][k]  (transposed on fill)
  const int t = threadIdx.x;
  const int lane = t & 31, laneLo = lane & 15, laneHi = lane >> 4;
  const int wid = t >> 5;
  const int waveM = wid & 1, waveN = wid >> 1;
  const int bM = blockIdx.y * 128, bN = blockIdx.x * 128;
  const v8f vzero = {0, 0, 0, 0, 0, 0, 0, 0};
  v8f acc[4][2];
  for (int i = 0; i < 4; ++i) for (int j = 0; j < 2; ++j) acc[i][j] = vzero;

  const int arow = t >> 1, acol = (t & 1) * 16;   // A tile: 16 halves/thread
  const int brow = t >> 3, bcol = (t & 7) * 16;   // B tile: 16 halves/thread
  const int aByte = (int)((((size_t)(bM + arow)) * K + acol) * sizeof(_Float16));
  const _Float16* bgBase = B + (size_t)brow * N + (bN + bcol);
  _Float16* asDst0 = &As[0][arow * 32 + acol];
  _Float16* asDst1 = &As[1][arow * 32 + acol];

  _Float16 breg[16];
  // ---- prologue: tile 0
  async_b128(A, aByte, asDst0);
  async_b128(A, aByte + 16, asDst0 + 8);
  *(uint4*)(breg)     = *(const uint4*)(bgBase);
  *(uint4*)(breg + 8) = *(const uint4*)(bgBase + 8);
#pragma unroll
  for (int j = 0; j < 16; ++j) BsT[0][(bcol + j) * 32 + brow] = breg[j];
  wait_async0();
  __syncthreads();

  const int nk = K >> 5;
  for (int k = 0; k < nk; ++k) {
    const int buf = k & 1;
    const int nbuf = buf ^ 1;
    const bool more = (k + 1) < nk;
    if (more) {
      const int kOff = ((k + 1) << 5) * 2;          // bytes along K
      _Float16* ad = nbuf ? asDst1 : asDst0;
      async_b128(A, aByte + kOff, ad);
      async_b128(A, aByte + kOff + 16, ad + 8);
      const _Float16* bg = bgBase + ((size_t)(k + 1) << 5) * N;
      *(uint4*)(breg)     = *(const uint4*)(bg);
      *(uint4*)(breg + 8) = *(const uint4*)(bg + 8);
    }

    v16h afr[4];
#pragma unroll
    for (int mt = 0; mt < 4; ++mt) {
      const _Float16* p = &As[buf][(waveM * 64 + mt * 16 + laneLo) * 32 + laneHi * 8];
      afr[mt] = ld_frag(p, p + 16);
    }
#pragma unroll
    for (int nt = 0; nt < 2; ++nt) {
      const _Float16* p = &BsT[buf][(waveN * 32 + nt * 16 + laneLo) * 32 + laneHi * 8];
      v16h bf = ld_frag(p, p + 16);
#pragma unroll
      for (int mt = 0; mt < 4; ++mt) acc[mt][nt] = wmma_f16(afr[mt], bf, acc[mt][nt]);
    }

    if (more) {
#pragma unroll
      for (int j = 0; j < 16; ++j) BsT[nbuf][(bcol + j) * 32 + brow] = breg[j];
      wait_async0();
    }
    __syncthreads();
  }

#pragma unroll
  for (int mt = 0; mt < 4; ++mt) {
#pragma unroll
    for (int nt = 0; nt < 2; ++nt) {
      const int col = bN + waveN * 32 + nt * 16 + laneLo;
      const int rowBase = bM + waveM * 64 + mt * 16 + laneHi * 8;
      const float bv = bias[col];
#pragma unroll
      for (int r = 0; r < 8; ++r) {
        float v = acc[mt][nt][r] + bv;
        const size_t idx = (size_t)(rowBase + r) * N + col;
        if (MODE == 0) {
          out16[idx] = (_Float16)v;
        } else if (MODE == 1) {
          float g = 0.5f * v * (1.0f + tanhf(0.79788456f * (v + 0.044715f * v * v * v)));
          out16[idx] = (_Float16)g;
        } else {
          outf[idx] = resid[idx] + v;
        }
      }
    }
  }
}

// ---------------------------------------------------------- flash attention
// One block (4 waves) per (64-query block, head). Bc=64, DH=64, f16 WMMA,
// online softmax in f32. Q and K tiles arrive by async global->LDS DMA
// (row-major == B^T layout for K already); V is VGPR-staged + transposed.
// qkv layout: [T, 3E] f16, out: [T, E] f16.
__global__ __launch_bounds__(128) void flash_attn(const _Float16* __restrict__ qkv,
                                                  _Float16* __restrict__ outp) {
  __shared__ _Float16 Qs[64 * 64];
  __shared__ _Float16 Ks[64 * 64];
  __shared__ _Float16 VsT[64 * 64];     // [d][kvrow]
  __shared__ _Float16 Ps[4 * 16 * 64];  // per-wave P strip [16][64]
  const int t = threadIdx.x;
  const int lane = t & 31, laneLo = lane & 15, laneHi = lane >> 4;
  const int w = t >> 5;
  const int hh = blockIdx.y;
  const int qBase = blockIdx.x * 64;
  const int RS = 3 * EMB;
  const int r2 = t >> 1, c2 = (t & 1) * 32;
  const int qByte = (int)((((size_t)(qBase + r2)) * RS + hh * DHEAD + c2) * 2);
  const int kColBase = EMB + hh * DHEAD;
  const _Float16* Vg = qkv + 2 * EMB + hh * DHEAD;

  // async copy Q tile (consumed after first in-loop barrier)
#pragma unroll
  for (int j = 0; j < 4; ++j)
    async_b128(qkv, qByte + j * 16, Qs + r2 * 64 + c2 + j * 8);

  const v8f vzero = {0, 0, 0, 0, 0, 0, 0, 0};
  v8f o[4];
  for (int i = 0; i < 4; ++i) o[i] = vzero;
  float mrow[8], lrow[8];
  for (int r = 0; r < 8; ++r) { mrow[r] = -3.0e30f; lrow[r] = 0.0f; }

  for (int kv = 0; kv < qBase + 64; kv += 64) {
    __syncthreads();
    {
      const int kByte = (int)((((size_t)(kv + r2)) * RS + kColBase + c2) * 2);
#pragma unroll
      for (int j = 0; j < 4; ++j)
        async_b128(qkv, kByte + j * 16, Ks + r2 * 64 + c2 + j * 8);
      const _Float16* vs = Vg + (size_t)(kv + r2) * RS + c2;
      _Float16 tmp[32];
#pragma unroll
      for (int j = 0; j < 4; ++j) *(uint4*)(tmp + j * 8) = *(const uint4*)(vs + j * 8);
#pragma unroll
      for (int j = 0; j < 32; ++j) VsT[(c2 + j) * 64 + r2] = tmp[j];
    }
    wait_async0();
    __syncthreads();

    // S = Q * K^T  (K rows are d-contiguous == B^T layout already)
    v8f s[4];
    for (int nt = 0; nt < 4; ++nt) s[nt] = vzero;
#pragma unroll
    for (int kc = 0; kc < 2; ++kc) {
      const _Float16* qp = Qs + (w * 16 + laneLo) * 64 + kc * 32 + laneHi * 8;
      v16h af = ld_frag(qp, qp + 16);
#pragma unroll
      for (int nt = 0; nt < 4; ++nt) {
        const _Float16* kp = Ks + (nt * 16 + laneLo) * 64 + kc * 32 + laneHi * 8;
        v16h bf = ld_frag(kp, kp + 16);
        s[nt] = wmma_f16(af, bf, s[nt]);
      }
    }

    // scale, causal mask, online softmax (row = r + laneHi*8, col = laneLo)
#pragma unroll
    for (int r = 0; r < 8; ++r) {
      const int qrow = qBase + w * 16 + r + laneHi * 8;
      float tmax = -3.0e30f;
#pragma unroll
      for (int nt = 0; nt < 4; ++nt) {
        const int kcol = kv + nt * 16 + laneLo;
        float sv = s[nt][r] * 0.125f;               // 1/sqrt(64)
        if (kcol > qrow) sv = -10000.0f;            // MASKED_BIAS
        s[nt][r] = sv;
        tmax = fmaxf(tmax, sv);
      }
#pragma unroll
      for (int d = 1; d < 16; d <<= 1) tmax = fmaxf(tmax, __shfl_xor(tmax, d, 32));
      const float mnew = fmaxf(mrow[r], tmax);
      const float alpha = __expf(mrow[r] - mnew);
      float rsum = 0.0f;
#pragma unroll
      for (int nt = 0; nt < 4; ++nt) {
        float p = __expf(s[nt][r] - mnew);
        s[nt][r] = p;
        rsum += p;
      }
#pragma unroll
      for (int d = 1; d < 16; d <<= 1) rsum += __shfl_xor(rsum, d, 32);
      lrow[r] = lrow[r] * alpha + rsum;
      mrow[r] = mnew;
#pragma unroll
      for (int nt = 0; nt < 4; ++nt) o[nt][r] *= alpha;
    }

    // C-layout -> A-layout for P via LDS
#pragma unroll
    for (int nt = 0; nt < 4; ++nt)
#pragma unroll
      for (int r = 0; r < 8; ++r)
        Ps[w * 1024 + (r + laneHi * 8) * 64 + nt * 16 + laneLo] = (_Float16)s[nt][r];
    __syncthreads();

    // O += P * V
#pragma unroll
    for (int kc = 0; kc < 2; ++kc) {
      const _Float16* pp = Ps + w * 1024 + laneLo * 64 + kc * 32 + laneHi * 8;
      v16h af = ld_frag(pp, pp + 16);
#pragma unroll
      for (int nt = 0; nt < 4; ++nt) {
        const _Float16* vp = VsT + (nt * 16 + laneLo) * 64 + kc * 32 + laneHi * 8;
        v16h bf = ld_frag(vp, vp + 16);
        o[nt] = wmma_f16(af, bf, o[nt]);
      }
    }
  }

#pragma unroll
  for (int nt = 0; nt < 4; ++nt)
#pragma unroll
    for (int r = 0; r < 8; ++r) {
      const int row = qBase + w * 16 + r + laneHi * 8;
      const int col = hh * DHEAD + nt * 16 + laneLo;
      outp[(size_t)row * EMB + col] = (_Float16)(o[nt][r] / lrow[r]);
    }
}

// ------------------------------------------------------------------ driver
extern "C" void kernel_launch(void* const* d_in, const int* in_sizes, int n_in,
                              void* d_out, int out_size, void* d_ws, size_t ws_size,
                              hipStream_t stream) {
  const float* x      = (const float*)d_in[0];
  const float* ln1_w  = (const float*)d_in[1];
  const float* ln1_b  = (const float*)d_in[2];
  const float* attn_w = (const float*)d_in[3];
  const float* attn_b = (const float*)d_in[4];
  const float* proj_w = (const float*)d_in[5];
  const float* proj_b = (const float*)d_in[6];
  const float* ln2_w  = (const float*)d_in[7];
  const float* ln2_b  = (const float*)d_in[8];
  const float* fc_w   = (const float*)d_in[9];
  const float* fc_b   = (const float*)d_in[10];
  const float* fc2_w  = (const float*)d_in[11];
  const float* fc2_b  = (const float*)d_in[12];
  float* h = (float*)d_out;  // residual stream lives in d_out

  char* ws = (char*)d_ws;
  _Float16* y16    = (_Float16*)(ws);                        //  4 MB ln out
  _Float16* qkv16  = (_Float16*)(ws + (size_t)( 4u << 20));  // 12 MB qkv
  _Float16* attn16 = (_Float16*)(ws + (size_t)(16u << 20));  //  4 MB attn out
  _Float16* m16    = (_Float16*)(ws + (size_t)(20u << 20));  // 16 MB mlp mid
  _Float16* aw16   = (_Float16*)(ws + (size_t)(36u << 20));  //  6 MB attn_w
  _Float16* pw16   = (_Float16*)(ws + (size_t)(42u << 20));  //  2 MB proj_w
  _Float16* fw16   = (_Float16*)(ws + (size_t)(44u << 20));  //  8 MB fc_w
  _Float16* f2w16  = (_Float16*)(ws + (size_t)(52u << 20));  //  8 MB fc2_w

  const int TE = T_SEQ * EMB;
  copy_f32<<<TE / 1024, 256, 0, stream>>>(x, h, TE);

  for (int l = 0; l < NL; ++l) {
    const int nAW = EMB * 3 * EMB, nPW = EMB * EMB, nFW = EMB * 4 * EMB;
    cvt_f32_f16<<<nAW / 1024, 256, 0, stream>>>(attn_w + (size_t)l * nAW, aw16, nAW);
    cvt_f32_f16<<<nPW / 1024, 256, 0, stream>>>(proj_w + (size_t)l * nPW, pw16, nPW);
    cvt_f32_f16<<<nFW / 1024, 256, 0, stream>>>(fc_w  + (size_t)l * nFW, fw16, nFW);
    cvt_f32_f16<<<nFW / 1024, 256, 0, stream>>>(fc2_w + (size_t)l * nFW, f2w16, nFW);

    ln_f16<<<T_SEQ, 256, 0, stream>>>(h, ln1_w + l * EMB, ln1_b + l * EMB, y16);
    gemm_wmma<0><<<dim3(3 * EMB / 128, T_SEQ / 128), 256, 0, stream>>>(
        y16, aw16, attn_b + (size_t)l * 3 * EMB, qkv16, nullptr, nullptr,
        T_SEQ, 3 * EMB, EMB);
    flash_attn<<<dim3(T_SEQ / 64, NH), 128, 0, stream>>>(qkv16, attn16);
    gemm_wmma<2><<<dim3(EMB / 128, T_SEQ / 128), 256, 0, stream>>>(
        attn16, pw16, proj_b + (size_t)l * EMB, nullptr, h, h, T_SEQ, EMB, EMB);

    ln_f16<<<T_SEQ, 256, 0, stream>>>(h, ln2_w + l * EMB, ln2_b + l * EMB, y16);
    gemm_wmma<1><<<dim3(4 * EMB / 128, T_SEQ / 128), 256, 0, stream>>>(
        y16, fw16, fc_b + (size_t)l * 4 * EMB, m16, nullptr, nullptr,
        T_SEQ, 4 * EMB, EMB);
    gemm_wmma<2><<<dim3(EMB / 128, T_SEQ / 128), 256, 0, stream>>>(
        m16, f2w16, fc2_b + (size_t)l * EMB, nullptr, h, h, T_SEQ, EMB, 4 * EMB);
  }
  (void)in_sizes; (void)n_in; (void)out_size; (void)ws_size;
}
